// MMMambaEncoderLayer_77206332113506
// MI455X (gfx1250) — matrix operand order repaired and checked by reading, not compiled, not run on hardware
//
#include <hip/hip_runtime.h>
#include <hip/hip_bf16.h>

// ---------------- problem constants ----------------
#define DM   512      // D_MODEL
#define DI   2048     // D_INNER
#define DS   16       // D_STATE
#define DR   32       // DT_RANK
#define BB   2        // BATCH
#define LL   512      // SEQ
#define BL   (BB*LL)  // 1024 rows

typedef __bf16 bf16_t;
typedef __attribute__((ext_vector_type(16))) __bf16 v16bf;
typedef __attribute__((ext_vector_type(8)))  __bf16 v8bf;
typedef __attribute__((ext_vector_type(8)))  float  v8f;

// ---------------- helpers ----------------
__device__ __forceinline__ v16bf load_a_frag(const bf16_t* base) {
  // 16 contiguous bf16: K = half*16 + 0..15
  v8bf lo = *(const v8bf*)(base);
  v8bf hi = *(const v8bf*)(base + 8);
  v16bf r;
#pragma unroll
  for (int i = 0; i < 8; ++i) { r[i] = lo[i]; r[i + 8] = hi[i]; }
  return r;
}
__device__ __forceinline__ v16bf load_b_frag(const bf16_t* base) {
  // 16 contiguous bf16: lane n=lane&15, K = half*16 + 0..15
  v8bf lo = *(const v8bf*)(base);
  v8bf hi = *(const v8bf*)(base + 8);
  v16bf r;
#pragma unroll
  for (int i = 0; i < 8; ++i) { r[i] = lo[i]; r[i + 8] = hi[i]; }
  return r;
}

// ---------------- f32 -> bf16 convert ----------------
__global__ void __launch_bounds__(256) k_f32_to_bf16(const float* __restrict__ in,
                                                     bf16_t* __restrict__ out, int n) {
  int i = blockIdx.x * 256 + threadIdx.x;
  if (i < n) out[i] = (bf16_t)in[i];
}

// ---------------- WMMA GEMM: C[M,N] = A[M,K] @ W[N,K]^T (bf16 in, f32 out) --------
// one wave per block; each wave computes a 16x64 output tile (4 WMMAs / K-step)
__global__ void __launch_bounds__(32) k_gemm_xwt(const bf16_t* __restrict__ A,
                                                 const bf16_t* __restrict__ W,
                                                 float* __restrict__ C,
                                                 int M, int N, int K) {
  const int lane = threadIdx.x;
  const int half = lane >> 4;
  const int mr   = lane & 15;
  const int m0   = blockIdx.x * 16;
  const int n0   = blockIdx.y * 64;

  const bf16_t* arow = A + (size_t)(m0 + mr) * K + half * 16;
  const bf16_t* w0 = W + (size_t)(n0 +  0 + mr) * K + half * 16;
  const bf16_t* w1 = W + (size_t)(n0 + 16 + mr) * K + half * 16;
  const bf16_t* w2 = W + (size_t)(n0 + 32 + mr) * K + half * 16;
  const bf16_t* w3 = W + (size_t)(n0 + 48 + mr) * K + half * 16;

  v8f c0 = {}, c1 = {}, c2 = {}, c3 = {};
  for (int k = 0; k < K; k += 32) {
    v16bf a  = load_a_frag(arow + k);
    v16bf b0 = load_b_frag(w0 + k);
    v16bf b1 = load_b_frag(w1 + k);
    v16bf b2 = load_b_frag(w2 + k);
    v16bf b3 = load_b_frag(w3 + k);
    c0 = __builtin_amdgcn_wmma_f32_16x16x32_bf16(false, a, false, b0, (short)0, c0, false, false);
    c1 = __builtin_amdgcn_wmma_f32_16x16x32_bf16(false, a, false, b1, (short)0, c1, false, false);
    c2 = __builtin_amdgcn_wmma_f32_16x16x32_bf16(false, a, false, b2, (short)0, c2, false, false);
    c3 = __builtin_amdgcn_wmma_f32_16x16x32_bf16(false, a, false, b3, (short)0, c3, false, false);
  }
  // D layout: lane n = mr, VGPR r -> row m = half*8 + r
#pragma unroll
  for (int r = 0; r < 8; ++r) {
    size_t off = (size_t)(m0 + half * 8 + r) * N + n0;
    C[off +  0 + mr] = c0[r];
    C[off + 16 + mr] = c1[r];
    C[off + 32 + mr] = c2[r];
    C[off + 48 + mr] = c3[r];
  }
}

// ---------------- causal depthwise conv (k=4) + SiLU; rev => flipped index domain ----
__global__ void __launch_bounds__(256) k_conv_silu(const float* __restrict__ xz,
                                                   const float* __restrict__ cw,
                                                   const float* __restrict__ cb,
                                                   float* __restrict__ xcf,
                                                   bf16_t* __restrict__ xch, int rev) {
  int i = blockIdx.x * 256 + threadIdx.x;
  if (i >= BB * LL * DI) return;
  int d = i % DI;
  int l = (i / DI) % LL;
  int b = i / (DI * LL);
  float acc = cb[d];
#pragma unroll
  for (int j = 0; j < 4; ++j) {
    int lf = l - 3 + j;
    if (lf >= 0) {
      int lo = rev ? (LL - 1 - lf) : lf;
      acc += xz[(size_t)(b * LL + lo) * (2 * DI) + d] * cw[d * 4 + j];
    }
  }
  float s = acc / (1.f + __expf(-acc));     // silu
  size_t o = (size_t)(b * LL + l) * DI + d;
  xcf[o] = s;
  xch[o] = (bf16_t)s;
}

// ---------------- split dbc -> dt_rank(bf16), B(f32), C(f32) ----------------
__global__ void __launch_bounds__(256) k_split_dbc(const float* __restrict__ dbc,
                                                   bf16_t* __restrict__ dtr,
                                                   float* __restrict__ Bm,
                                                   float* __restrict__ Cm) {
  int i = blockIdx.x * 256 + threadIdx.x;
  if (i >= BL * 64) return;
  int row = i >> 6, c = i & 63;
  float v = dbc[i];
  if (c < DR)            dtr[row * DR + c] = (bf16_t)v;
  else if (c < DR + DS)  Bm[row * DS + (c - DR)] = v;
  else                   Cm[row * DS + (c - DR - DS)] = v;
}

// ---------------- softplus(x + bias), in-place capable ----------------
__global__ void __launch_bounds__(256) k_softplus_bias(const float* __restrict__ in,
                                                       const float* __restrict__ bdt,
                                                       float* __restrict__ out) {
  int i = blockIdx.x * 256 + threadIdx.x;
  if (i >= BL * DI) return;
  float x = in[i] + bdt[i % DI];
  out[i] = (x > 20.f) ? x : logf(1.f + __expf(x));
}

// ---------------- selective-scan: h[16] in regs, B/C chunk-staged in LDS ----------
#define SCAN_CHUNK 64
__global__ void __launch_bounds__(256) k_ssm_scan(const float* __restrict__ u,
                                                  const float* __restrict__ dt,
                                                  const float* __restrict__ alog,
                                                  const float* __restrict__ Bm,
                                                  const float* __restrict__ Cm,
                                                  const float* __restrict__ dd,
                                                  const float* __restrict__ xz,
                                                  float* __restrict__ comb,
                                                  int rev, int accum) {
  const int b = blockIdx.x / (DI / 256);
  const int d = (blockIdx.x % (DI / 256)) * 256 + threadIdx.x;

  float Ac[DS];
#pragma unroll
  for (int n = 0; n < DS; ++n) Ac[n] = -__expf(alog[d * DS + n]);
  float h[DS];
#pragma unroll
  for (int n = 0; n < DS; ++n) h[n] = 0.f;
  const float ddv = dd[d];

  __shared__ float sB[SCAN_CHUNK * DS];
  __shared__ float sC[SCAN_CHUNK * DS];

  for (int l0 = 0; l0 < LL; l0 += SCAN_CHUNK) {
    __syncthreads();
    for (int i = threadIdx.x; i < SCAN_CHUNK * DS; i += 256) {
      sB[i] = Bm[(size_t)(b * LL + l0) * DS + i];
      sC[i] = Cm[(size_t)(b * LL + l0) * DS + i];
    }
    __syncthreads();
    for (int lc = 0; lc < SCAN_CHUNK; ++lc) {
      const int l = l0 + lc;
      const size_t o = (size_t)(b * LL + l) * DI + d;
      const float dtv = dt[o];
      const float uv  = u[o];
      float y = 0.f;
#pragma unroll
      for (int n = 0; n < DS; ++n) {
        float dA = __expf(dtv * Ac[n]);
        h[n] = dA * h[n] + (dtv * sB[lc * DS + n]) * uv;
        y += h[n] * sC[lc * DS + n];
      }
      y += uv * ddv;
      const int lo = rev ? (LL - 1 - l) : l;            // unflip on store
      const float zv = xz[(size_t)(b * LL + lo) * (2 * DI) + DI + d];
      const float res = y * (zv / (1.f + __expf(-zv))); // gate by silu(z)
      const size_t oo = (size_t)(b * LL + lo) * DI + d;
      if (accum) comb[oo] += res; else comb[oo] = res;
    }
  }
}

// ---------------- LayerNorm + residual (one row per block) ----------------
__global__ void __launch_bounds__(256) k_ln_residual(const float* __restrict__ y,
                                                     const float* __restrict__ xin,
                                                     const float* __restrict__ g,
                                                     const float* __restrict__ bb,
                                                     float* __restrict__ out) {
  const int row = blockIdx.x;
  const int t = threadIdx.x;
  const float* yr = y + (size_t)row * DM;
  float v0 = yr[t], v1 = yr[t + 256];
  float s = v0 + v1, q = v0 * v0 + v1 * v1;
#pragma unroll
  for (int off = 16; off > 0; off >>= 1) {  // wave32 reduce
    s += __shfl_down(s, off, 32);
    q += __shfl_down(q, off, 32);
  }
  __shared__ float ws_[8], wq_[8];
  __shared__ float mu_s, rs_s;
  if ((t & 31) == 0) { ws_[t >> 5] = s; wq_[t >> 5] = q; }
  __syncthreads();
  if (t == 0) {
    float S = 0.f, Q = 0.f;
    for (int i = 0; i < 8; ++i) { S += ws_[i]; Q += wq_[i]; }
    float mu = S / DM;
    mu_s = mu;
    rs_s = rsqrtf(Q / DM - mu * mu + 1e-6f);
  }
  __syncthreads();
  const float mu = mu_s, rs = rs_s;
  size_t o = (size_t)row * DM;
  out[o + t]       = xin[o + t]       + (v0 - mu) * rs * g[t]       + bb[t];
  out[o + t + 256] = xin[o + t + 256] + (v1 - mu) * rs * g[t + 256] + bb[t + 256];
}

// ---------------- host orchestration ----------------
static inline size_t align256(size_t x) { return (x + 255) & ~(size_t)255; }

extern "C" void kernel_launch(void* const* d_in, const int* in_sizes, int n_in,
                              void* d_out, int out_size, void* d_ws, size_t ws_size,
                              hipStream_t stream) {
  const float* e_x    = (const float*)d_in[0];
  const float* g_x    = (const float*)d_in[1];
  const float* win    = (const float*)d_in[2];
  const float* wout   = (const float*)d_in[3];
  const float* convw_f= (const float*)d_in[4];
  const float* convb_f= (const float*)d_in[5];
  const float* wx_f   = (const float*)d_in[6];
  const float* wdt_f  = (const float*)d_in[7];
  const float* bdt_f  = (const float*)d_in[8];
  const float* alog_f = (const float*)d_in[9];
  const float* dd_f   = (const float*)d_in[10];
  const float* convw_b= (const float*)d_in[11];
  const float* convb_b= (const float*)d_in[12];
  const float* wx_b   = (const float*)d_in[13];
  const float* wdt_b  = (const float*)d_in[14];
  const float* bdt_b  = (const float*)d_in[15];
  const float* alog_b = (const float*)d_in[16];
  const float* dd_b   = (const float*)d_in[17];
  const float* ln1g   = (const float*)d_in[18];
  const float* ln1b   = (const float*)d_in[19];
  const float* ln2g   = (const float*)d_in[20];
  const float* ln2b   = (const float*)d_in[21];
  float* outf = (float*)d_out;

  // bump allocator over workspace (~61 MB total)
  char* ws = (char*)d_ws;
  size_t off = 0;
  auto alloc = [&](size_t bytes) { void* p = ws + off; off = align256(off + bytes); return p; };

  bf16_t* winh  = (bf16_t*)alloc((size_t)2 * DI * DM * 2);   // 4096x512
  bf16_t* wouth = (bf16_t*)alloc((size_t)DM * DI * 2);       // 512x2048
  bf16_t* wxfh  = (bf16_t*)alloc((size_t)64 * DI * 2);
  bf16_t* wxbh  = (bf16_t*)alloc((size_t)64 * DI * 2);
  bf16_t* wdtfh = (bf16_t*)alloc((size_t)DI * DR * 2);
  bf16_t* wdtbh = (bf16_t*)alloc((size_t)DI * DR * 2);
  bf16_t* xh    = (bf16_t*)alloc((size_t)BL * DM * 2);
  float*  xzf   = (float*) alloc((size_t)BL * 2 * DI * 4);
  float*  xcf   = (float*) alloc((size_t)BL * DI * 4);
  bf16_t* xch   = (bf16_t*)alloc((size_t)BL * DI * 2);
  float*  dbcf  = (float*) alloc((size_t)BL * 64 * 4);
  bf16_t* dtrh  = (bf16_t*)alloc((size_t)BL * DR * 2);
  float*  Bmf   = (float*) alloc((size_t)BL * DS * 4);
  float*  Cmf   = (float*) alloc((size_t)BL * DS * 4);
  float*  dtf   = (float*) alloc((size_t)BL * DI * 4);
  float*  combf = (float*) alloc((size_t)BL * DI * 4);
  bf16_t* combh = (bf16_t*)alloc((size_t)BL * DI * 2);
  float*  y1f   = (float*) alloc((size_t)BL * DM * 4);

  auto cvt = [&](const float* src, bf16_t* dst, int n) {
    k_f32_to_bf16<<<(n + 255) / 256, 256, 0, stream>>>(src, dst, n);
  };

  // weights -> bf16 once (shared by both streams)
  cvt(win,   winh,  2 * DI * DM);
  cvt(wout,  wouth, DM * DI);
  cvt(wx_f,  wxfh,  64 * DI);
  cvt(wx_b,  wxbh,  64 * DI);
  cvt(wdt_f, wdtfh, DI * DR);
  cvt(wdt_b, wdtbh, DI * DR);

  struct Branch {
    const float *cw, *cb, *bdt, *alog, *dd;
    const bf16_t *wxh, *wdth;
    int rev;
  };
  Branch brf{convw_f, convb_f, bdt_f, alog_f, dd_f, wxfh, wdtfh, 0};
  Branch brb{convw_b, convb_b, bdt_b, alog_b, dd_b, wxbh, wdtbh, 1};

  auto run_stream = [&](const float* x, const float* lg, const float* lb, float* outp) {
    // in-proj: xz = x @ win^T   (M=1024,N=4096,K=512)
    cvt(x, xh, BL * DM);
    k_gemm_xwt<<<dim3(BL / 16, (2 * DI) / 64), 32, 0, stream>>>(xh, winh, xzf, BL, 2 * DI, DM);

    for (int dir = 0; dir < 2; ++dir) {
      const Branch& br = dir ? brb : brf;
      k_conv_silu<<<(BB * LL * DI + 255) / 256, 256, 0, stream>>>(
          xzf, br.cw, br.cb, xcf, xch, br.rev);
      // x-proj: dbc = xc @ wx^T  (M=1024,N=64,K=2048)
      k_gemm_xwt<<<dim3(BL / 16, 1), 32, 0, stream>>>(xch, br.wxh, dbcf, BL, 64, DI);
      k_split_dbc<<<(BL * 64 + 255) / 256, 256, 0, stream>>>(dbcf, dtrh, Bmf, Cmf);
      // dt: dtr @ wdt^T (M=1024,N=2048,K=32), then softplus(+bias) in-place
      k_gemm_xwt<<<dim3(BL / 16, DI / 64), 32, 0, stream>>>(dtrh, br.wdth, dtf, BL, DI, DR);
      k_softplus_bias<<<(BL * DI + 255) / 256, 256, 0, stream>>>(dtf, br.bdt, dtf);
      // scan (fwd overwrites comb, bwd accumulates; both unflip+gate on store)
      k_ssm_scan<<<BB * (DI / 256), 256, 0, stream>>>(
          xcf, dtf, br.alog, Bmf, Cmf, br.dd, xzf, combf, br.rev, dir);
    }

    // out-proj: y1 = comb @ wout^T (M=1024,N=512,K=2048)
    cvt(combf, combh, BL * DI);
    k_gemm_xwt<<<dim3(BL / 16, DM / 64), 32, 0, stream>>>(combh, wouth, y1f, BL, DM, DI);
    // LN + residual
    k_ln_residual<<<BL, 256, 0, stream>>>(y1f, x, lg, lb, outp);
  };

  run_stream(e_x, ln1g, ln1b, outf);
  run_stream(g_x, ln2g, ln2b, outf + (size_t)BL * DM);
}
